// ConceptLearner_GRU_As_MT_36524401885491
// MI455X (gfx1250) — matrix operand order repaired and checked by reading, not compiled, not run on hardware
//
#include <hip/hip_runtime.h>
#include <hip/hip_bf16.h>
#include <math.h>

// ---------------------------------------------------------------------------
// ConceptLearner GRU seq2seq, CDNA5 (gfx1250) WMMA implementation.
//
// dims: B=128, S=512, D=256, H=512, E=256, V=1002, T=34.
//
//  * fp16 data / fp32 accumulate via v_wmma_f32_16x16x32_f16 (wave32).
//  * Input GEMMs (x@Wih^T) batched over all timesteps (no recurrence).
//  * Recurrent GEMM (h@Whh^T) in a persistent 12-WG kernel (one WG per WGP):
//      - Whh slice (128 KB) staged in LDS once via async global->LDS copies,
//      - h (128 KB) staged in LDS once per step (4x L2 traffic reduction),
//      - atomic grid barrier between GEMM phase and gate phase per step.
//  * B-fragment N-guard is compile-time (only FC3 with V=1002 pays for it).
// ---------------------------------------------------------------------------

typedef __attribute__((ext_vector_type(16))) _Float16 v16h;
typedef __attribute__((ext_vector_type(8)))  _Float16 v8h;
typedef __attribute__((ext_vector_type(8)))  float    v8f;

// async global->LDS builtin operand types (per hipcc diagnostic: int4 ptrs)
typedef __attribute__((__vector_size__(4 * sizeof(int)))) int vsi4;
typedef __attribute__((address_space(1))) vsi4 g_vsi4;
typedef __attribute__((address_space(3))) vsi4 l_vsi4;

#define DEV __device__ __forceinline__

static constexpr int Bsz = 128;
static constexpr int Hsz = 512;
static constexpr int Ssz = 512;
static constexpr int Tsz = 34;
static constexpr int Esz = 256;
static constexpr int Dsz = 256;
static constexpr int Vsz = 1002;
static constexpr int G3H = 3 * Hsz;      // 1536
static constexpr int GRU_WGS = 12;       // 12 * 128 cols == 1536
static constexpr int HL_HALVES = Bsz * Hsz;          // 65536 fp16 = 128 KB
static constexpr size_t GRU_LDS_BYTES = 2u * HL_HALVES * sizeof(_Float16); // 256 KB

DEV v16h cat8(v8h lo, v8h hi) {
  return __builtin_shufflevector(lo, hi, 0,1,2,3,4,5,6,7,8,9,10,11,12,13,14,15);
}

// A fragment: 16x32 fp16, row-major source (lda in halves).
// lanes 0-15: M=lane, K runs [0..7],[16..23]; lanes 16-31 shift K by +8.
DEV v16h load_a_frag(const _Float16* __restrict__ A, int lda, int m0, int k0) {
  int lane = threadIdx.x & 31;
  int hi8  = (lane >> 4) << 3;                       // 0 or 8 halves
  const _Float16* p = A + (size_t)(m0 + (lane & 15)) * lda + k0 + hi8;
  return cat8(*(const v8h*)p, *(const v8h*)(p + 16));
}

// B fragment: 32x16 fp16 (K x N) with B[k][n] = W[n][k], W row-major [N,K].
// lane: column n = n0+lane%16, K runs [0..7],[16..23] shifted by +8 for lanes 16-31.
template <bool GUARD>
DEV v16h load_b_frag(const _Float16* __restrict__ W, int ldw, int n0, int k0,
                     int nvalid) {
  int lane = threadIdx.x & 31;
  int n = n0 + (lane & 15);
  if (GUARD && n >= nvalid) { v16h z = {}; return z; }  // only FC3 (V=1002)
  int kk = k0 + ((lane >> 4) << 3);
  const _Float16* p = W + (size_t)n * ldw + kk;
  return cat8(*(const v8h*)p, *(const v8h*)(p + 16));
}

DEV v8f wmma_f16(v16h a, v16h b, v8f c) {
  return __builtin_amdgcn_wmma_f32_16x16x32_f16(false, a, false, b, (short)0, c,
                                                false, false);
}

// C/D 16x16 f32 layout: VGPR r -> lanes 0-15 M=r, lanes 16-31 M=r+8, N=lane%16.
template <typename T>
DEV void store_c(T* __restrict__ C, int ldc, int m0, int n0, const v8f& acc,
                 const float* __restrict__ bias, int nvalid) {
  int lane = threadIdx.x & 31;
  int col = n0 + (lane & 15);
  if (col >= nvalid) return;
  int rbase = m0 + ((lane >> 4) << 3);
  float bv = bias ? bias[col] : 0.0f;
#pragma unroll
  for (int r = 0; r < 8; ++r)
    C[(size_t)(rbase + r) * ldc + col] = (T)(acc[r] + bv);
}

// ---------------------------------------------------------------------------
// Async global->LDS copy (16 B per lane-op), ASYNCcnt-tracked on CDNA5.
// Falls back to a plain LDS store where the builtin is unavailable.
// ---------------------------------------------------------------------------
DEV void cp_async_b128(void* lds_dst, const void* gsrc) {
#if __has_builtin(__builtin_amdgcn_global_load_async_to_lds_b128)
  __builtin_amdgcn_global_load_async_to_lds_b128(
      (g_vsi4*)gsrc, (l_vsi4*)lds_dst, 0, 0);
#else
  *(v8h*)lds_dst = *(const v8h*)gsrc;
#endif
}

DEV void cp_async_wait_all() {
#if __has_builtin(__builtin_amdgcn_global_load_async_to_lds_b128)
#if __has_builtin(__builtin_amdgcn_s_wait_asynccnt)
  __builtin_amdgcn_s_wait_asynccnt(0);
#else
  asm volatile("s_wait_asynccnt 0" ::: "memory");
#endif
#endif
  __syncthreads();
}

// ---------------------------------------------------------------------------
// Generic GEMM: C[M,N] = A[M,K] * W[N,K]^T + bias.  256 thr = 8 waves.
// Workgroup tile 128x128; wave tile 64x32 (4x2 WMMA tiles, 8 accumulators).
// grid.x = ceil(N/128), grid.y = M/128 (all M here are multiples of 128).
// ---------------------------------------------------------------------------
template <typename TOut, bool GUARD>
__global__ __launch_bounds__(256) void gemm_f16(
    const _Float16* __restrict__ A, const _Float16* __restrict__ W,
    const float* __restrict__ bias, TOut* __restrict__ C, int K, int Nvalid,
    int lda, int ldw, int ldc) {
  int wave = threadIdx.x >> 5;
  int m0 = blockIdx.y * 128 + (wave >> 2) * 64;
  int n0 = blockIdx.x * 128 + (wave & 3) * 32;
  v8f acc[4][2] = {};
  for (int k = 0; k < K; k += 32) {
    v16h a0 = load_a_frag(A, lda, m0 +  0, k);
    v16h a1 = load_a_frag(A, lda, m0 + 16, k);
    v16h a2 = load_a_frag(A, lda, m0 + 32, k);
    v16h a3 = load_a_frag(A, lda, m0 + 48, k);
    v16h b0 = load_b_frag<GUARD>(W, ldw, n0 +  0, k, Nvalid);
    v16h b1 = load_b_frag<GUARD>(W, ldw, n0 + 16, k, Nvalid);
    acc[0][0] = wmma_f16(a0, b0, acc[0][0]);
    acc[0][1] = wmma_f16(a0, b1, acc[0][1]);
    acc[1][0] = wmma_f16(a1, b0, acc[1][0]);
    acc[1][1] = wmma_f16(a1, b1, acc[1][1]);
    acc[2][0] = wmma_f16(a2, b0, acc[2][0]);
    acc[2][1] = wmma_f16(a2, b1, acc[2][1]);
    acc[3][0] = wmma_f16(a3, b0, acc[3][0]);
    acc[3][1] = wmma_f16(a3, b1, acc[3][1]);
  }
#pragma unroll
  for (int i = 0; i < 4; ++i)
#pragma unroll
    for (int j = 0; j < 2; ++j)
      store_c(C, ldc, m0 + i * 16, n0 + j * 16, acc[i][j], bias, Nvalid);
}

// ---------------------------------------------------------------------------
// Grid-wide barrier for the persistent GRU kernel (monotonic counter).
// ---------------------------------------------------------------------------
DEV void grid_barrier(unsigned* cnt, unsigned nwg, unsigned phase) {
  __threadfence();
  __syncthreads();
  if (threadIdx.x == 0) {
    atomicAdd(cnt, 1u);
    while (atomicAdd(cnt, 0u) < nwg * phase) __builtin_amdgcn_s_sleep(2);
  }
  __syncthreads();
}

// ---------------------------------------------------------------------------
// Persistent GRU layer: S sequential steps, 12 WGs (one 128-col slice of 3H
// each, pinned one-per-WGP by the 256 KB LDS footprint).
//   prologue: Whh slice (128 KB) -> LDS (async, once)
//   phase 1 : h (128 KB) -> LDS (async), gh = h @ Whh^T + bhh (WMMA from LDS)
//   phase 2 : gates elementwise, update h (fp16 global), emit y_t
// GI[B*S, 3H] fp16 precomputed (= x@Wih^T + bih), rows (b*S + t).
// ---------------------------------------------------------------------------
__global__ __launch_bounds__(256) void gru_layer(
    const _Float16* __restrict__ GI, const _Float16* __restrict__ Whh,
    const float* __restrict__ bhh, _Float16* __restrict__ hbuf,
    _Float16* __restrict__ ybuf, float* __restrict__ gh,
    unsigned* __restrict__ cnt, int S) {
  extern __shared__ __align__(16) _Float16 smem[];
  _Float16* HL = smem;               // [128][512] current h
  _Float16* WL = smem + HL_HALVES;   // [128][512] this WG's Whh rows
  const unsigned NWG = gridDim.x;    // 12
  unsigned phase = 0;
  int wave = threadIdx.x >> 5;
  int m0  = (wave >> 2) * 64;                      // B=128 -> single M tile
  int n0l = (wave & 3) * 32;                       // local col inside slice
  int nbase = blockIdx.x * 128;                    // this WG's 3H columns

  // stage Whh rows [nbase, nbase+128) once: 8192 x 16 B
  {
    const _Float16* wsrc = Whh + (size_t)nbase * Hsz;
    for (int i = threadIdx.x; i < 8192; i += 256)
      cp_async_b128(WL + i * 8, wsrc + i * 8);
    cp_async_wait_all();
  }

  for (int t = 0; t < S; ++t) {
    // ---- phase 1: stage h, recurrent GEMM out of LDS ----
    for (int i = threadIdx.x; i < 8192; i += 256)
      cp_async_b128(HL + i * 8, hbuf + i * 8);
    cp_async_wait_all();

    v8f acc[4][2] = {};
    for (int k = 0; k < Hsz; k += 32) {
      v16h a0 = load_a_frag(HL, Hsz, m0 +  0, k);
      v16h a1 = load_a_frag(HL, Hsz, m0 + 16, k);
      v16h a2 = load_a_frag(HL, Hsz, m0 + 32, k);
      v16h a3 = load_a_frag(HL, Hsz, m0 + 48, k);
      v16h b0 = load_b_frag<false>(WL, Hsz, n0l +  0, k, G3H);
      v16h b1 = load_b_frag<false>(WL, Hsz, n0l + 16, k, G3H);
      acc[0][0] = wmma_f16(a0, b0, acc[0][0]);
      acc[0][1] = wmma_f16(a0, b1, acc[0][1]);
      acc[1][0] = wmma_f16(a1, b0, acc[1][0]);
      acc[1][1] = wmma_f16(a1, b1, acc[1][1]);
      acc[2][0] = wmma_f16(a2, b0, acc[2][0]);
      acc[2][1] = wmma_f16(a2, b1, acc[2][1]);
      acc[3][0] = wmma_f16(a3, b0, acc[3][0]);
      acc[3][1] = wmma_f16(a3, b1, acc[3][1]);
    }
#pragma unroll
    for (int i = 0; i < 4; ++i)
#pragma unroll
      for (int j = 0; j < 2; ++j)
        store_c(gh, G3H, m0 + i * 16, nbase + n0l + j * 16, acc[i][j], bhh, G3H);
    ++phase;
    grid_barrier(cnt, NWG, phase);

    // ---- phase 2: gates ----
    for (int idx = blockIdx.x * blockDim.x + threadIdx.x; idx < Bsz * Hsz;
         idx += (int)(NWG * blockDim.x)) {
      int b = idx >> 9, j = idx & (Hsz - 1);
      const _Float16* gi = GI + ((size_t)b * S + t) * G3H;
      float ir = (float)gi[j], iz = (float)gi[Hsz + j], in = (float)gi[2 * Hsz + j];
      const float* ghr = gh + (size_t)b * G3H;
      float hr = ghr[j], hz = ghr[Hsz + j], hn = ghr[2 * Hsz + j];
      float r = 1.f / (1.f + __expf(-(ir + hr)));
      float z = 1.f / (1.f + __expf(-(iz + hz)));
      float n = tanhf(in + r * hn);
      float hp = (float)hbuf[idx];
      float hnew = (1.f - z) * n + z * hp;
      hbuf[idx] = (_Float16)hnew;
      ybuf[((size_t)b * S + t) * Hsz + j] = (_Float16)hnew;
    }
    ++phase;
    grid_barrier(cnt, NWG, phase);
  }
}

// ---------------------------------------------------------------------------
// Small helper kernels
// ---------------------------------------------------------------------------
__global__ void zero_u32(unsigned* __restrict__ p, int n) {
  int i = blockIdx.x * blockDim.x + threadIdx.x;
  if (i < n) p[i] = 0u;
}

__global__ void cvt_f16(const float* __restrict__ x, _Float16* __restrict__ y,
                        size_t n) {
  for (size_t i = blockIdx.x * (size_t)blockDim.x + threadIdx.x; i < n;
       i += (size_t)gridDim.x * blockDim.x)
    y[i] = (_Float16)x[i];
}

__global__ void embed_f16(const int* __restrict__ ids,
                          const float* __restrict__ emb,
                          _Float16* __restrict__ out, int rows, int E) {
  size_t n = (size_t)rows * E;
  for (size_t i = blockIdx.x * (size_t)blockDim.x + threadIdx.x; i < n;
       i += (size_t)gridDim.x * blockDim.x) {
    size_t r = i / E, c = i % E;
    out[i] = (_Float16)emb[(size_t)ids[r] * E + c];
  }
}

// pooled[b,j] = sum_{t < len[b]} y[b,t,j]
__global__ void pool_masked(const _Float16* __restrict__ y,
                            const int* __restrict__ len,
                            _Float16* __restrict__ pooled, int T) {
  for (int idx = blockIdx.x * blockDim.x + threadIdx.x; idx < Bsz * Hsz;
       idx += gridDim.x * blockDim.x) {
    int b = idx >> 9, j = idx & (Hsz - 1);
    int L = len[b]; L = L > T ? T : L;
    float s = 0.f;
    for (int t = 0; t < L; ++t) s += (float)y[((size_t)b * T + t) * Hsz + j];
    pooled[idx] = (_Float16)s;
  }
}

// loss = -mean_b log_softmax(sigmoid(x))[b, V-1];  C3 row stride 1024.
__global__ __launch_bounds__(128) void loss_kernel(const float* __restrict__ C3,
                                                   float* __restrict__ out) {
  __shared__ float red[128];
  int b = threadIdx.x;
  const float* row = C3 + (size_t)b * 1024;
  float m = -3.4e38f;
  for (int v = 0; v < Vsz; ++v) {
    float s = 1.f / (1.f + __expf(-row[v]));
    m = fmaxf(m, s);
  }
  float sum = 0.f;
  for (int v = 0; v < Vsz; ++v) {
    float s = 1.f / (1.f + __expf(-row[v]));
    sum += __expf(s - m);
  }
  float sEnd = 1.f / (1.f + __expf(-row[Vsz - 1]));
  red[b] = -(sEnd - m - __logf(sum));
  __syncthreads();
  for (int off = 64; off > 0; off >>= 1) {
    if (b < off) red[b] += red[b + off];
    __syncthreads();
  }
  if (b == 0) out[0] = red[0] / (float)Bsz;
}

// ---------------------------------------------------------------------------
// Host-side orchestration
// ---------------------------------------------------------------------------
extern "C" void kernel_launch(void* const* d_in, const int* in_sizes, int n_in,
                              void* d_out, int out_size, void* d_ws,
                              size_t ws_size, hipStream_t stream) {
  (void)in_sizes; (void)n_in; (void)out_size; (void)ws_size;
  const float* source   = (const float*)d_in[0];
  const int*   tgt_ids  = (const int*)d_in[1];
  const int*   lengths  = (const int*)d_in[2];
  const float* emb      = (const float*)d_in[3];
  const float* eWih0 = (const float*)d_in[4],  *eWhh0 = (const float*)d_in[5];
  const float* ebih0 = (const float*)d_in[6],  *ebhh0 = (const float*)d_in[7];
  const float* eWih1 = (const float*)d_in[8],  *eWhh1 = (const float*)d_in[9];
  const float* ebih1 = (const float*)d_in[10], *ebhh1 = (const float*)d_in[11];
  const float* dWih0 = (const float*)d_in[12], *dWhh0 = (const float*)d_in[13];
  const float* dbih0 = (const float*)d_in[14], *dbhh0 = (const float*)d_in[15];
  const float* dWih1 = (const float*)d_in[16], *dWhh1 = (const float*)d_in[17];
  const float* dbih1 = (const float*)d_in[18], *dbhh1 = (const float*)d_in[19];
  const float* fcW1 = (const float*)d_in[20], *fcb1 = (const float*)d_in[21];
  const float* fcW2 = (const float*)d_in[22], *fcb2 = (const float*)d_in[23];
  const float* fcW3 = (const float*)d_in[24], *fcb3 = (const float*)d_in[25];

  // bump allocator over d_ws (256 B aligned); ~350 MB total.
  size_t off = 0;
  auto alloc = [&](size_t bytes) -> void* {
    void* p = (char*)d_ws + off;
    off += (bytes + 255) & ~(size_t)255;
    return p;
  };
  unsigned*  cnt   = (unsigned*)alloc(256);                      // 4 barriers
  _Float16*  hA    = (_Float16*)alloc((size_t)Bsz * Hsz * 2);    // layer-0 h
  _Float16*  hB    = (_Float16*)alloc((size_t)Bsz * Hsz * 2);    // layer-1 h
  float*     gh    = (float*)alloc((size_t)Bsz * G3H * 4);
  _Float16*  src16 = (_Float16*)alloc((size_t)Bsz * Ssz * Dsz * 2);
  _Float16* eWih0h = (_Float16*)alloc((size_t)G3H * Dsz * 2);
  _Float16* eWhh0h = (_Float16*)alloc((size_t)G3H * Hsz * 2);
  _Float16* eWih1h = (_Float16*)alloc((size_t)G3H * Hsz * 2);
  _Float16* eWhh1h = (_Float16*)alloc((size_t)G3H * Hsz * 2);
  _Float16* dWih0h = (_Float16*)alloc((size_t)G3H * Esz * 2);
  _Float16* dWhh0h = (_Float16*)alloc((size_t)G3H * Hsz * 2);
  _Float16* dWih1h = (_Float16*)alloc((size_t)G3H * Hsz * 2);
  _Float16* dWhh1h = (_Float16*)alloc((size_t)G3H * Hsz * 2);
  _Float16* fcW1h  = (_Float16*)alloc((size_t)5 * Hsz * Hsz * 2);
  _Float16* fcW2h  = (_Float16*)alloc((size_t)2 * Hsz * 5 * Hsz * 2);
  _Float16* fcW3h  = (_Float16*)alloc((size_t)Vsz * 2 * Hsz * 2);
  _Float16* GIenc  = (_Float16*)alloc((size_t)Bsz * Ssz * G3H * 2);  // 201 MB
  _Float16* yenc0  = (_Float16*)alloc((size_t)Bsz * Ssz * Hsz * 2);  //  67 MB
  _Float16* tg16   = (_Float16*)alloc((size_t)Bsz * Tsz * Esz * 2);
  _Float16* GIdec  = (_Float16*)alloc((size_t)Bsz * Tsz * G3H * 2);
  _Float16* ydec0  = (_Float16*)alloc((size_t)Bsz * Tsz * Hsz * 2);
  _Float16* ydec1  = (_Float16*)alloc((size_t)Bsz * Tsz * Hsz * 2);
  _Float16* pooled = (_Float16*)alloc((size_t)Bsz * Hsz * 2);
  _Float16* a1     = (_Float16*)alloc((size_t)Bsz * 5 * Hsz * 2);
  _Float16* a2     = (_Float16*)alloc((size_t)Bsz * 2 * Hsz * 2);
  float*    C3     = (float*)alloc((size_t)Bsz * 1024 * 4);

  // 0) zero barrier counters + h0 states (contiguous region at ws start)
  int zn = (int)((256 + 2 * (size_t)Bsz * Hsz * 2 + 3) / 4);
  zero_u32<<<(zn + 255) / 256, 256, 0, stream>>>((unsigned*)d_ws, zn);

  // 1) fp32 -> fp16 conversions
  auto cvt = [&](const float* x, _Float16* y, size_t n) {
    int g = (int)((n + 255) / 256); if (g > 2048) g = 2048;
    cvt_f16<<<g, 256, 0, stream>>>(x, y, n);
  };
  cvt(source, src16, (size_t)Bsz * Ssz * Dsz);
  cvt(eWih0, eWih0h, (size_t)G3H * Dsz);
  cvt(eWhh0, eWhh0h, (size_t)G3H * Hsz);
  cvt(eWih1, eWih1h, (size_t)G3H * Hsz);
  cvt(eWhh1, eWhh1h, (size_t)G3H * Hsz);
  cvt(dWih0, dWih0h, (size_t)G3H * Esz);
  cvt(dWhh0, dWhh0h, (size_t)G3H * Hsz);
  cvt(dWih1, dWih1h, (size_t)G3H * Hsz);
  cvt(dWhh1, dWhh1h, (size_t)G3H * Hsz);
  cvt(fcW1, fcW1h, (size_t)5 * Hsz * Hsz);
  cvt(fcW2, fcW2h, (size_t)2 * Hsz * 5 * Hsz);
  cvt(fcW3, fcW3h, (size_t)Vsz * 2 * Hsz);
  embed_f16<<<1024, 256, 0, stream>>>(tgt_ids, emb, tg16, Bsz * Tsz, Esz);

  // ---- encoder ----
  gemm_f16<_Float16, false><<<dim3(12, (Bsz * Ssz) / 128), 256, 0, stream>>>(
      src16, eWih0h, ebih0, GIenc, Dsz, G3H, Dsz, Dsz, G3H);
  gru_layer<<<GRU_WGS, 256, GRU_LDS_BYTES, stream>>>(GIenc, eWhh0h, ebhh0, hA,
                                                     yenc0, gh, cnt + 0, Ssz);
  gemm_f16<_Float16, false><<<dim3(12, (Bsz * Ssz) / 128), 256, 0, stream>>>(
      yenc0, eWih1h, ebih1, GIenc, Hsz, G3H, Hsz, Hsz, G3H);
  gru_layer<<<GRU_WGS, 256, GRU_LDS_BYTES, stream>>>(GIenc, eWhh1h, ebhh1, hB,
                                                     yenc0, gh, cnt + 1, Ssz);

  // ---- decoder (h0 = encoder final h, already resident in hA/hB) ----
  gemm_f16<_Float16, false><<<dim3(12, (Bsz * Tsz) / 128), 256, 0, stream>>>(
      tg16, dWih0h, dbih0, GIdec, Esz, G3H, Esz, Esz, G3H);
  gru_layer<<<GRU_WGS, 256, GRU_LDS_BYTES, stream>>>(GIdec, dWhh0h, dbhh0, hA,
                                                     ydec0, gh, cnt + 2, Tsz);
  gemm_f16<_Float16, false><<<dim3(12, (Bsz * Tsz) / 128), 256, 0, stream>>>(
      ydec0, dWih1h, dbih1, GIdec, Hsz, G3H, Hsz, Hsz, G3H);
  gru_layer<<<GRU_WGS, 256, GRU_LDS_BYTES, stream>>>(GIdec, dWhh1h, dbhh1, hB,
                                                     ydec1, gh, cnt + 3, Tsz);

  // ---- masked time pooling + FC head ----
  pool_masked<<<256, 256, 0, stream>>>(ydec1, lengths, pooled, Tsz);
  gemm_f16<_Float16, false><<<dim3(20, 1), 256, 0, stream>>>(
      pooled, fcW1h, fcb1, a1, Hsz, 5 * Hsz, Hsz, Hsz, 5 * Hsz);
  gemm_f16<_Float16, false><<<dim3(8, 1), 256, 0, stream>>>(
      a1, fcW2h, fcb2, a2, 5 * Hsz, 2 * Hsz, 5 * Hsz, 5 * Hsz, 2 * Hsz);
  // FC3: [128,1024] x [1024,1002] (ragged N -> guarded B loads, ldc = 1024)
  gemm_f16<float, true><<<dim3(8, 1), 256, 0, stream>>>(
      a2, fcW3h, fcb3, C3, 2 * Hsz, Vsz, 2 * Hsz, 2 * Hsz, 1024);

  loss_kernel<<<1, 128, 0, stream>>>(C3, (float*)d_out);
}
